// EmotionDetection_33406255628697
// MI455X (gfx1250) — matrix operand order repaired and checked by reading, not compiled, not run on hardware
//
#include <hip/hip_runtime.h>
#include <hip/hip_bf16.h>
#include <stdint.h>

// ---------------------------------------------------------------------------
// EmotionDetection (DialogueRNN-style) for MI455X / gfx1250, wave32 + WMMA.
// Round 4: same 4-kernel/step structure as round 3; codegen fixes:
//   * __launch_bounds__(128, 1) on WMMA kernels (grid is only 32-64 blocks,
//     so trade occupancy for VGPRs -> pipelined loads, fewer hazard NOPs).
//   * k_gru_g K-loop split into (ih + q) and (hh) passes to shrink the
//     per-iteration live fragment set.
//   * #pragma unroll 2 on k_gru2 K-loops.
// ---------------------------------------------------------------------------

constexpr int BB = 64;    // batch
constexpr int TT = 256;   // seq len
constexpr int EE = 1024;  // embed
constexpr int HH = 512;   // hidden (emotion GRU)

typedef __attribute__((ext_vector_type(16))) __bf16 bf16x16;
typedef __attribute__((ext_vector_type(8)))  float  floatx8;

union Frag { uint32_t u[8]; bf16x16 v; };

// ---- A fragment (16x32, M x K), CDNA5 wave32 layout -----------------------
__device__ __forceinline__ bf16x16 load_a(const __bf16* __restrict__ src, int lda) {
    const int lane = threadIdx.x & 31;
    const int m    = lane & 15;
    const int kh   = (lane >> 4) * 8;
    const __bf16* p = src + (size_t)m * lda;
    Frag f;
#pragma unroll
    for (int j = 0; j < 4; ++j) {
        f.u[j]     = *(const uint32_t*)(p + kh + 2 * j);
        f.u[j + 4] = *(const uint32_t*)(p + 16 + kh + 2 * j);
    }
    return f.v;
}

// ---- B fragment (32x16, K x N); weight stored [N_total, K] bf16 -----------
__device__ __forceinline__ bf16x16 load_b(const __bf16* __restrict__ Wt, int ldb) {
    const int lane = threadIdx.x & 31;
    const int n    = lane & 15;
    const int kb   = (lane >> 4) * 8;
    const __bf16* p = Wt + (size_t)n * ldb + kb;
    Frag f;
#pragma unroll
    for (int j = 0; j < 4; ++j) {
        f.u[j]     = *(const uint32_t*)(p + 2 * j);
        f.u[j + 4] = *(const uint32_t*)(p + 16 + 2 * j);
    }
    return f.v;
}

// Packed bf16 add of two fragments (layouts match elementwise).
__device__ __forceinline__ bf16x16 frag_add(bf16x16 a, bf16x16 b) {
    Frag fa, fb, fd;
    fa.v = a; fb.v = b;
#pragma unroll
    for (int j = 0; j < 8; ++j)
        asm("v_pk_add_bf16 %0, %1, %2" : "=v"(fd.u[j]) : "v"(fa.u[j]), "v"(fb.u[j]));
    return fd.v;
}

#define WMMA_BF16(A, Bm, C) \
    __builtin_amdgcn_wmma_f32_16x16x32_bf16(false, (A), false, (Bm), (short)0, (C), false, false)

__device__ __forceinline__ float sigmoidf_(float x) { return 1.f / (1.f + __expf(-x)); }

// ---------------------------------------------------------------------------
// Kernel 1: global-state GRU + q GEMM.
//   gi = (u_t + party) @ Wih^T ; gh = hist_i @ Whh^T ; q = u_t @ Watt
// Grid EE/16 blocks (n0), block 128 = 4 waves (one 16-row batch tile each).
// ---------------------------------------------------------------------------
__global__ void __launch_bounds__(128, 1)
k_gru_g(const __bf16* __restrict__ ub_t,      // ub + t*EE, stride TT*EE
        const __bf16* __restrict__ party_bf,  // [64, EE]
        const __bf16* __restrict__ hist_bf_i, // [64, EE]
        const float*  __restrict__ hist_i,    // fp32 hprev
        const __bf16* __restrict__ Wih,       // [3EE, EE]
        const __bf16* __restrict__ Whh,       // [3EE, EE]
        const __bf16* __restrict__ Watt,      // [EE, EE] ([n,k])
        const float*  __restrict__ bih,
        const float*  __restrict__ bhh,
        float*  __restrict__ hist_ip1,        // [64, EE] fp32
        __bf16* __restrict__ hist_bf_ip1,     // [64, EE] bf16 mirror
        float*  __restrict__ qout)            // [64, EE]
{
    const int lane = threadIdx.x & 31;
    const int m0   = (threadIdx.x >> 5) * 16;
    const int n0   = blockIdx.x * 16;

    floatx8 ai0 = {}, ai1 = {}, ai2 = {};
    floatx8 ah0 = {}, ah1 = {}, ah2 = {};
    floatx8 aq  = {};

    // Pass 1: input-side gates + attention query.
    for (int k0 = 0; k0 < EE; k0 += 32) {
        bf16x16 au = load_a(ub_t + (size_t)m0 * TT * EE + k0, TT * EE);
        bf16x16 ap = load_a(party_bf + (size_t)m0 * EE + k0, EE);
        bf16x16 ax = frag_add(au, ap);   // x = u_t + party (packed bf16)

        bf16x16 bi0 = load_b(Wih + ((size_t)(0 * EE + n0)) * EE + k0, EE);
        bf16x16 bi1 = load_b(Wih + ((size_t)(1 * EE + n0)) * EE + k0, EE);
        bf16x16 bi2 = load_b(Wih + ((size_t)(2 * EE + n0)) * EE + k0, EE);
        bf16x16 bq  = load_b(Watt + (size_t)n0 * EE + k0, EE);

        ai0 = WMMA_BF16(ax, bi0, ai0);
        ai1 = WMMA_BF16(ax, bi1, ai1);
        ai2 = WMMA_BF16(ax, bi2, ai2);
        aq  = WMMA_BF16(au, bq,  aq);
    }
    // Pass 2: hidden-side gates.
    for (int k0 = 0; k0 < EE; k0 += 32) {
        bf16x16 ah  = load_a(hist_bf_i + (size_t)m0 * EE + k0, EE);
        bf16x16 bh0 = load_b(Whh + ((size_t)(0 * EE + n0)) * EE + k0, EE);
        bf16x16 bh1 = load_b(Whh + ((size_t)(1 * EE + n0)) * EE + k0, EE);
        bf16x16 bh2 = load_b(Whh + ((size_t)(2 * EE + n0)) * EE + k0, EE);
        ah0 = WMMA_BF16(ah, bh0, ah0);
        ah1 = WMMA_BF16(ah, bh1, ah1);
        ah2 = WMMA_BF16(ah, bh2, ah2);
    }

    const int n = n0 + (lane & 15);
    const float bir = bih[n],           bhr = bhh[n];
    const float biz = bih[EE + n],      bhz = bhh[EE + n];
    const float bin_ = bih[2 * EE + n], bhn = bhh[2 * EE + n];
#pragma unroll
    for (int v = 0; v < 8; ++v) {
        const int m = m0 + v + ((lane >> 4) << 3);
        const float r  = sigmoidf_(ai0[v] + bir + ah0[v] + bhr);
        const float z  = sigmoidf_(ai1[v] + biz + ah1[v] + bhz);
        const float nn = tanhf(ai2[v] + bin_ + r * (ah2[v] + bhn));
        const float hp = hist_i[(size_t)m * EE + n];
        const float hnew = (1.f - z) * nn + z * hp;
        hist_ip1[(size_t)m * EE + n]    = hnew;
        hist_bf_ip1[(size_t)m * EE + n] = (__bf16)hnew;
        qout[(size_t)m * EE + n]        = aq[v];
    }
}

// ---------------------------------------------------------------------------
// Kernel 2: fused attention. One block per batch element b (64 blocks, 256T).
// ---------------------------------------------------------------------------
__global__ void k_attn(const float* __restrict__ q,
                       const float* __restrict__ hist,
                       const float* __restrict__ u,
                       __bf16* __restrict__ xp_bf,
                       int nk, int t)
{
    __shared__ float sw[256];
    __shared__ float red[256];
    const int b    = blockIdx.x;
    const int tid  = threadIdx.x;
    const int lane = tid & 31;
    const int wave = tid >> 5;
    const float* qrow = q + (size_t)b * EE;

    for (int k = wave; k < nk; k += 8) {
        const float* hrow = hist + ((size_t)k * BB + b) * EE;
        float s = 0.f;
        for (int e = lane; e < EE; e += 32) s += qrow[e] * hrow[e];
#pragma unroll
        for (int off = 16; off > 0; off >>= 1) s += __shfl_xor(s, off, 32);
        if (lane == 0) sw[k] = s;
    }
    __syncthreads();

    const float v = (tid < nk) ? sw[tid] : -3.4e38f;
    red[tid] = v;
    __syncthreads();
    for (int s = 128; s > 0; s >>= 1) {
        if (tid < s) red[tid] = fmaxf(red[tid], red[tid + s]);
        __syncthreads();
    }
    const float mx = red[0];
    __syncthreads();
    const float ev = (tid < nk) ? __expf(v - mx) : 0.f;
    red[tid] = ev;
    __syncthreads();
    for (int s = 128; s > 0; s >>= 1) {
        if (tid < s) red[tid] += red[tid + s];
        __syncthreads();
    }
    const float inv = 1.f / red[0];
    __syncthreads();
    sw[tid] = ev * inv;
    __syncthreads();

    for (int e = tid; e < EE; e += 256) {
        float c = 0.f;
        for (int k = 0; k < nk; ++k)
            c += sw[k] * hist[((size_t)k * BB + b) * EE + e];
        xp_bf[(size_t)b * EE + e] =
            (__bf16)(u[((size_t)b * TT + t) * EE + e] + c);
    }
}

// ---------------------------------------------------------------------------
// Kernel 3/4: generic GRU cell. bf16 A inputs; fp32 hprev; writes fp32 state,
// bf16 mirror, and optional strided copy (d_out).
// ---------------------------------------------------------------------------
__global__ void __launch_bounds__(128, 1)
k_gru2(const __bf16* __restrict__ xb, int ldx,  // [64, Kx]
       const __bf16* __restrict__ hb, int ldh,  // [64, Kh]
       const __bf16* __restrict__ Wih,          // [3N, Kx]
       const __bf16* __restrict__ Whh,          // [3N, Kh]
       const float*  __restrict__ bih,
       const float*  __restrict__ bhh,
       const float*  __restrict__ hprev,        // [64, N] fp32
       float*  __restrict__ out1,               // [64, N]
       __bf16* __restrict__ out1_bf,            // [64, N] mirror
       float*  __restrict__ out2, int out2_stride,
       int N, int Kx, int Kh)
{
    const int lane = threadIdx.x & 31;
    const int m0   = (threadIdx.x >> 5) * 16;
    const int n0   = blockIdx.x * 16;

    floatx8 ai0 = {}, ai1 = {}, ai2 = {};
    floatx8 ah0 = {}, ah1 = {}, ah2 = {};

#pragma unroll 2
    for (int k0 = 0; k0 < Kx; k0 += 32) {
        bf16x16 a  = load_a(xb + (size_t)m0 * ldx + k0, ldx);
        bf16x16 b0 = load_b(Wih + ((size_t)(0 * N + n0)) * Kx + k0, Kx);
        bf16x16 b1 = load_b(Wih + ((size_t)(1 * N + n0)) * Kx + k0, Kx);
        bf16x16 b2 = load_b(Wih + ((size_t)(2 * N + n0)) * Kx + k0, Kx);
        ai0 = WMMA_BF16(a, b0, ai0);
        ai1 = WMMA_BF16(a, b1, ai1);
        ai2 = WMMA_BF16(a, b2, ai2);
    }
#pragma unroll 2
    for (int k0 = 0; k0 < Kh; k0 += 32) {
        bf16x16 a  = load_a(hb + (size_t)m0 * ldh + k0, ldh);
        bf16x16 b0 = load_b(Whh + ((size_t)(0 * N + n0)) * Kh + k0, Kh);
        bf16x16 b1 = load_b(Whh + ((size_t)(1 * N + n0)) * Kh + k0, Kh);
        bf16x16 b2 = load_b(Whh + ((size_t)(2 * N + n0)) * Kh + k0, Kh);
        ah0 = WMMA_BF16(a, b0, ah0);
        ah1 = WMMA_BF16(a, b1, ah1);
        ah2 = WMMA_BF16(a, b2, ah2);
    }

    const int n = n0 + (lane & 15);
    const float bir = bih[n],          bhr = bhh[n];
    const float biz = bih[N + n],      bhz = bhh[N + n];
    const float bin_ = bih[2 * N + n], bhn = bhh[2 * N + n];
#pragma unroll
    for (int v = 0; v < 8; ++v) {
        const int m = m0 + v + ((lane >> 4) << 3);
        const float r  = sigmoidf_(ai0[v] + bir + ah0[v] + bhr);
        const float z  = sigmoidf_(ai1[v] + biz + ah1[v] + bhz);
        const float nn = tanhf(ai2[v] + bin_ + r * (ah2[v] + bhn));
        const float hp = hprev[(size_t)m * N + n];
        const float hnew = (1.f - z) * nn + z * hp;
        out1[(size_t)m * N + n]    = hnew;
        out1_bf[(size_t)m * N + n] = (__bf16)hnew;
        if (out2) out2[(size_t)m * out2_stride + n] = hnew;
    }
}

// fp32 -> bf16 flat conversion.
__global__ void k_cvt(const float* __restrict__ in, __bf16* __restrict__ out, int n)
{
    const int i = blockIdx.x * blockDim.x + threadIdx.x;
    if (i < n) out[i] = (__bf16)in[i];
}

// fp32 [E,E] -> bf16 transposed [n,k] (for w_att).
__global__ void k_cvt_t(const float* __restrict__ in, __bf16* __restrict__ out)
{
    const int i = blockIdx.x * blockDim.x + threadIdx.x;
    const int n = i >> 10;
    const int k = i & (EE - 1);
    out[i] = (__bf16)in[(size_t)k * EE + n];
}

// ---------------------------------------------------------------------------
extern "C" void kernel_launch(void* const* d_in, const int* in_sizes, int n_in,
                              void* d_out, int out_size, void* d_ws, size_t ws_size,
                              hipStream_t stream)
{
    (void)in_sizes; (void)n_in; (void)out_size; (void)ws_size;
    const float* u     = (const float*)d_in[0];
    const float* wg_ih = (const float*)d_in[1];
    const float* wg_hh = (const float*)d_in[2];
    const float* bg_ih = (const float*)d_in[3];
    const float* bg_hh = (const float*)d_in[4];
    const float* w_att = (const float*)d_in[5];
    const float* wp_ih = (const float*)d_in[6];
    const float* wp_hh = (const float*)d_in[7];
    const float* bp_ih = (const float*)d_in[8];
    const float* bp_hh = (const float*)d_in[9];
    const float* we_ih = (const float*)d_in[10];
    const float* we_hh = (const float*)d_in[11];
    const float* be_ih = (const float*)d_in[12];
    const float* be_hh = (const float*)d_in[13];
    float* out = (float*)d_out;

    // ---- workspace carve-out (aligned 256B) -------------------------------
    char* wsp = (char*)d_ws;
    auto alloc = [&](size_t bytes) -> void* {
        void* p = (void*)wsp;
        wsp += (bytes + 255) & ~(size_t)255;
        return p;
    };
    float*  hist    = (float*) alloc((size_t)(TT + 1) * BB * EE * sizeof(float));
    __bf16* hist_bf = (__bf16*)alloc((size_t)(TT + 1) * BB * EE * 2);
    // party ping-pong: [parity*2 + buf]; buf alternates each visit of a parity
    float*  psb[4];
    __bf16* psb_bf[4];
    for (int i = 0; i < 4; ++i) {
        psb[i]    = (float*) alloc((size_t)BB * EE * sizeof(float));
        psb_bf[i] = (__bf16*)alloc((size_t)BB * EE * 2);
    }
    float*  emob[2];
    __bf16* emob_bf[2];
    for (int i = 0; i < 2; ++i) {
        emob[i]    = (float*) alloc((size_t)BB * HH * sizeof(float));
        emob_bf[i] = (__bf16*)alloc((size_t)BB * HH * 2);
    }
    float*  q     = (float*) alloc((size_t)BB * EE * sizeof(float));
    __bf16* xp_bf = (__bf16*)alloc((size_t)BB * EE * 2);
    __bf16* ub    = (__bf16*)alloc((size_t)BB * TT * EE * 2);
    __bf16* wgihb = (__bf16*)alloc((size_t)3 * EE * EE * 2);
    __bf16* wghhb = (__bf16*)alloc((size_t)3 * EE * EE * 2);
    __bf16* wpihb = (__bf16*)alloc((size_t)3 * EE * EE * 2);
    __bf16* wphhb = (__bf16*)alloc((size_t)3 * EE * EE * 2);
    __bf16* wattb = (__bf16*)alloc((size_t)EE * EE * 2);
    __bf16* weihb = (__bf16*)alloc((size_t)3 * HH * EE * 2);
    __bf16* wehhb = (__bf16*)alloc((size_t)3 * HH * HH * 2);

    // ---- zero initial state (ws is poisoned each run) ---------------------
    hipMemsetAsync(hist,       0, (size_t)BB * EE * sizeof(float), stream); // slot 0
    hipMemsetAsync(hist_bf,    0, (size_t)BB * EE * 2, stream);             // slot 0
    hipMemsetAsync(psb[0],     0, (size_t)BB * EE * sizeof(float), stream);
    hipMemsetAsync(psb[2],     0, (size_t)BB * EE * sizeof(float), stream);
    hipMemsetAsync(psb_bf[0],  0, (size_t)BB * EE * 2, stream);
    hipMemsetAsync(psb_bf[2],  0, (size_t)BB * EE * 2, stream);
    hipMemsetAsync(emob[0],    0, (size_t)BB * HH * sizeof(float), stream);
    hipMemsetAsync(emob_bf[0], 0, (size_t)BB * HH * 2, stream);

    // ---- one-time bf16 conversions ----------------------------------------
    auto cvt = [&](const float* src, __bf16* dst, int n) {
        k_cvt<<<(n + 255) / 256, 256, 0, stream>>>(src, dst, n);
    };
    cvt(u,     ub,    BB * TT * EE);
    cvt(wg_ih, wgihb, 3 * EE * EE);
    cvt(wg_hh, wghhb, 3 * EE * EE);
    cvt(wp_ih, wpihb, 3 * EE * EE);
    cvt(wp_hh, wphhb, 3 * EE * EE);
    cvt(we_ih, weihb, 3 * HH * EE);
    cvt(we_hh, wehhb, 3 * HH * HH);
    k_cvt_t<<<(EE * EE + 255) / 256, 256, 0, stream>>>(w_att, wattb);

    // ---- sequential recurrence over T (4 kernels per step) ----------------
    for (int t = 0; t < TT; ++t) {
        const int p = t & 1;
        const int j = (t >> 1) & 1;
        float*  party_in     = psb[p * 2 + j];
        float*  party_out    = psb[p * 2 + (1 - j)];
        __bf16* party_in_bf  = psb_bf[p * 2 + j];
        __bf16* party_out_bf = psb_bf[p * 2 + (1 - j)];
        float*  emo_in       = emob[t & 1];
        float*  emo_out      = emob[1 - (t & 1)];
        __bf16* emo_in_bf    = emob_bf[t & 1];
        __bf16* emo_out_bf   = emob_bf[1 - (t & 1)];
        float*  hist_i       = hist + (size_t)t * BB * EE;
        float*  hist_ip1     = hist_i + (size_t)BB * EE;
        __bf16* hist_bf_i    = hist_bf + (size_t)t * BB * EE;
        __bf16* hist_bf_ip1  = hist_bf_i + (size_t)BB * EE;

        // global-state GRU + q GEMM
        k_gru_g<<<EE / 16, 128, 0, stream>>>(ub + (size_t)t * EE, party_in_bf,
                                             hist_bf_i, hist_i,
                                             wgihb, wghhb, wattb, bg_ih, bg_hh,
                                             hist_ip1, hist_bf_ip1, q);
        // attention over hist[0..t] -> xp_bf = bf16(u_t + ctx)
        k_attn<<<BB, 256, 0, stream>>>(q, hist, u, xp_bf, t + 1, t);
        // party GRU: party_out = GRU(xp, party_in)
        k_gru2<<<EE / 16, 128, 0, stream>>>(xp_bf, EE, party_in_bf, EE,
                                            wpihb, wphhb, bp_ih, bp_hh,
                                            party_in, party_out, party_out_bf,
                                            nullptr, 0, EE, EE, EE);
        // emotion GRU: emo_out = GRU(party_out, emo_in); emit d_out[:, t, :]
        k_gru2<<<HH / 16, 128, 0, stream>>>(party_out_bf, EE, emo_in_bf, HH,
                                            weihb, wehhb, be_ih, be_hh,
                                            emo_in, emo_out, emo_out_bf,
                                            out + (size_t)t * HH, TT * HH,
                                            HH, EE, HH);
    }
}
